// InfoNCELoss_76914274337243
// MI455X (gfx1250) — matrix operand order repaired and checked by reading, not compiled, not run on hardware
//
#include <hip/hip_runtime.h>
#include <hip/hip_bf16.h>

typedef __attribute__((ext_vector_type(16))) _Float16 v16h;
typedef __attribute__((ext_vector_type(8)))  _Float16 v8h;
typedef __attribute__((ext_vector_type(4)))  _Float16 v4h;
typedef __attribute__((ext_vector_type(8)))  float    v8f;

#define NROWS   8192
#define DIM     128
#define INV_TAU 5.0f                    // 1/0.2
#define L2E_TAU 7.2134752044448169f     // log2(e)/tau : exp(x/tau) = exp2(x*L2E_TAU)
#define EPSN    1e-12f
#define EXP5    148.4131591025766f      // exp(1/tau) = diag of self-Gram (f32-exact)
#define CSPLIT  8
#define CHUNK   (NROWS / CSPLIT)        // 1024 columns per block
#define WAVES   8
#define MROWS   32                      // rows per wave (2 x 16-row WMMA tiles)
#define WGROWS  (WAVES * MROWS)         // 256 rows per workgroup

// ---------------------------------------------------------------------------
// Kernel 1: row-wise L2 normalize (f32), emit f16 copies for WMMA, and
// compute pos_i = <h1n_i, h2n_i> in f32 (the precision-critical term).
// ---------------------------------------------------------------------------
__global__ __launch_bounds__(256) void norm_kernel(
    const float* __restrict__ h1, const float* __restrict__ h2,
    _Float16* __restrict__ h1n, _Float16* __restrict__ h2n,
    float* __restrict__ d12) {
  const int wave = threadIdx.x >> 5;
  const int lane = threadIdx.x & 31;
  const int row  = blockIdx.x * 8 + wave;
  const size_t base = (size_t)row * DIM + lane * 4;

  float4 x1 = *(const float4*)(h1 + base);
  float4 x2 = *(const float4*)(h2 + base);

  float ss1 = x1.x*x1.x + x1.y*x1.y + x1.z*x1.z + x1.w*x1.w;
  float ss2 = x2.x*x2.x + x2.y*x2.y + x2.z*x2.z + x2.w*x2.w;
  #pragma unroll
  for (int m = 1; m < 32; m <<= 1) {
    ss1 += __shfl_xor(ss1, m, 32);
    ss2 += __shfl_xor(ss2, m, 32);
  }
  const float s1 = 1.0f / fmaxf(sqrtf(ss1), EPSN);
  const float s2 = 1.0f / fmaxf(sqrtf(ss2), EPSN);

  float n1x = x1.x*s1, n1y = x1.y*s1, n1z = x1.z*s1, n1w = x1.w*s1;
  float n2x = x2.x*s2, n2y = x2.y*s2, n2z = x2.z*s2, n2w = x2.w*s2;

  v4h a, b;
  a[0]=(_Float16)n1x; a[1]=(_Float16)n1y; a[2]=(_Float16)n1z; a[3]=(_Float16)n1w;
  b[0]=(_Float16)n2x; b[1]=(_Float16)n2y; b[2]=(_Float16)n2z; b[3]=(_Float16)n2w;
  *(v4h*)(h1n + base) = a;
  *(v4h*)(h2n + base) = b;

  float d = n1x*n2x + n1y*n2y + n1z*n2z + n1w*n2w;
  #pragma unroll
  for (int m = 1; m < 32; m <<= 1) d += __shfl_xor(d, m, 32);
  if (lane == 0) d12[row] = d;
}

// ---------------------------------------------------------------------------
// Kernel 2: zero the rowsum accumulators.
// ---------------------------------------------------------------------------
__global__ void zero_kernel(float* __restrict__ p, int n) {
  int i = blockIdx.x * blockDim.x + threadIdx.x;
  if (i < n) p[i] = 0.0f;
}

// ---------------------------------------------------------------------------
// One 32x16 tile: load all 4 B fragments (8x b128 clause), run the 8-WMMA
// chain with decreasing loadcnt waits, then exp2 + row/col accumulation.
// Branch-free; COLSUM resolved at compile time.
// ---------------------------------------------------------------------------
template <bool COLSUM>
__device__ __forceinline__ void gram_tile(
    const _Float16* __restrict__ bp, const v16h (&af)[2][4],
    float (&acc)[2][8], float* __restrict__ slab, int slabIdx,
    int lp, int half) {
  v16h b0 = *(const v16h*)(bp);
  v16h b1 = *(const v16h*)(bp + 32);
  v16h b2 = *(const v16h*)(bp + 64);
  v16h b3 = *(const v16h*)(bp + 96);

  v8f c0 = {}, c1 = {};
  c0 = __builtin_amdgcn_wmma_f32_16x16x32_f16(false, af[0][0], false, b0, (short)0, c0, false, false);
  c1 = __builtin_amdgcn_wmma_f32_16x16x32_f16(false, af[1][0], false, b0, (short)0, c1, false, false);
  c0 = __builtin_amdgcn_wmma_f32_16x16x32_f16(false, af[0][1], false, b1, (short)0, c0, false, false);
  c1 = __builtin_amdgcn_wmma_f32_16x16x32_f16(false, af[1][1], false, b1, (short)0, c1, false, false);
  c0 = __builtin_amdgcn_wmma_f32_16x16x32_f16(false, af[0][2], false, b2, (short)0, c0, false, false);
  c1 = __builtin_amdgcn_wmma_f32_16x16x32_f16(false, af[1][2], false, b2, (short)0, c1, false, false);
  c0 = __builtin_amdgcn_wmma_f32_16x16x32_f16(false, af[0][3], false, b3, (short)0, c0, false, false);
  c1 = __builtin_amdgcn_wmma_f32_16x16x32_f16(false, af[1][3], false, b3, (short)0, c1, false, false);

  float colv = 0.0f;
  #pragma unroll
  for (int r = 0; r < 8; ++r) {
    float e0 = __builtin_amdgcn_exp2f(c0[r] * L2E_TAU);
    float e1 = __builtin_amdgcn_exp2f(c1[r] * L2E_TAU);
    acc[0][r] += e0;
    acc[1][r] += e1;
    colv += e0 + e1;
  }
  if (COLSUM) {
    colv += __shfl_xor(colv, 16, 32);            // full 32-row column sum
    if (half == 0) atomicAdd(&slab[slabIdx + lp], colv);   // ds_add_f32
  }
}

// ---------------------------------------------------------------------------
// Kernel 3: fused GEMM(A·Bᵀ) -> exp2 -> row/col sums.
//   blockIdx.z = m : 0:S11(sym)  1:S22(sym)  2:S12 (colsums -> S21 rows)
// Symmetric matrices: wave at 32-row block i0 sweeps only jb >= i0; the <=2
// diagonal-block tiles are rows-only, everything beyond mirrors into colsums.
// All wave-uniform values are forced into SGPRs (readfirstlane) so both loops
// are scalar-bounded and branch-free inside.
// ---------------------------------------------------------------------------
__global__ __launch_bounds__(256) void gram_kernel(
    const _Float16* __restrict__ h1n, const _Float16* __restrict__ h2n,
    float* __restrict__ R) {
  __shared__ float colslab[WAVES * CHUNK];   // 32 KB of 320 KB LDS

  const int m = blockIdx.z;
  const _Float16* __restrict__ A = (m == 1) ? h2n : h1n;
  const _Float16* __restrict__ B = (m == 0) ? h1n : h2n;
  const int rowDst = m;                      // R11 / R22 / R12
  const int colDst = (m == 2) ? 3 : m;       // R21 for S12, else self

  const int wave = __builtin_amdgcn_readfirstlane(threadIdx.x) >> 5;  // SGPR
  const int lane = threadIdx.x & 31;
  const int lp   = lane & 15;
  const int half = lane >> 4;
  const int i0   = blockIdx.y * WGROWS + wave * MROWS;                // SGPR

  float* myslab = colslab + wave * CHUNK;
  for (int c = lane; c < CHUNK; c += 32) myslab[c] = 0.0f;

  // Preload A fragments: 2 row-subtiles x 4 K-chunks (loop-invariant, 64 VGPRs)
  v16h afrag[2][4];
  #pragma unroll
  for (int s = 0; s < 2; ++s) {
    const _Float16* arow = A + (size_t)(i0 + s * 16 + lp) * DIM;
    #pragma unroll
    for (int t = 0; t < 4; ++t) {
      const int kk = t * 32;
      v8h lo = *(const v8h*)(arow + kk + half * 16);
      v8h hi = *(const v8h*)(arow + kk + half * 16 + 8);
      #pragma unroll
      for (int i = 0; i < 8; ++i) { afrag[s][t][i] = lo[i]; afrag[s][t][i + 8] = hi[i]; }
    }
  }

  const _Float16* bbase = B + (size_t)lp * DIM + half * 16;
  const int jBegin = blockIdx.x * CHUNK;
  const int jEnd   = jBegin + CHUNK;

  // Split ranges (all SGPR math):
  //   rows-only:  [jStart, jMid)  -- symmetric diagonal block, <=2 tiles
  //   rows+cols:  [jMid,   jEnd)
  int jStart, jMid;
  if (m < 2) {
    jStart = jBegin > i0 ? jBegin : i0;
    int d  = i0 + MROWS;
    jMid   = d > jBegin ? d : jBegin;
    jMid   = jMid < jEnd ? jMid : jEnd;
  } else {
    jStart = jBegin;
    jMid   = jBegin;
  }

  float acc[2][8];
  #pragma unroll
  for (int s = 0; s < 2; ++s)
    #pragma unroll
    for (int r = 0; r < 8; ++r) acc[s][r] = 0.0f;

  for (int jb = jStart; jb < jMid; jb += 16)   // diagonal-block tiles
    gram_tile<false>(bbase + (size_t)jb * DIM, afrag, acc, myslab, 0, lp, half);

  for (int jb = jMid; jb < jEnd; jb += 16)     // main sweep (rows + mirrored cols)
    gram_tile<true>(bbase + (size_t)jb * DIM, afrag, acc, myslab, jb - jBegin, lp, half);

  // Row sums: C layout -> VGPR r, half h holds row i0 + s*16 + 8*h + r.
  #pragma unroll
  for (int s = 0; s < 2; ++s) {
    #pragma unroll
    for (int r = 0; r < 8; ++r) {
      float v = acc[s][r];
      v += __shfl_xor(v, 1, 32);
      v += __shfl_xor(v, 2, 32);
      v += __shfl_xor(v, 4, 32);
      v += __shfl_xor(v, 8, 32);
      if (lp == 0)
        atomicAdd(&R[rowDst * NROWS + i0 + s * 16 + half * 8 + r], v);
    }
  }

  // Cross-wave column reduction, one global atomic per (touched) column.
  __syncthreads();
  for (int c = threadIdx.x; c < CHUNK; c += 256) {
    float s = 0.0f;
    #pragma unroll
    for (int w = 0; w < WAVES; ++w) s += colslab[w * CHUNK + c];
    if (s != 0.0f) atomicAdd(&R[colDst * NROWS + jBegin + c], s);
  }
}

// ---------------------------------------------------------------------------
// Kernel 4: final loss.
//   neg1_i = R11_i - e^5 + R12_i ; neg2_i = R22_i - e^5 + R21_i
//   loss_i = -d12_i/tau + 0.5*(log neg1 + log neg2);  out = mean(loss)
// ---------------------------------------------------------------------------
__global__ __launch_bounds__(256) void loss_kernel(
    const float* __restrict__ R, const float* __restrict__ d12,
    float* __restrict__ out) {
  __shared__ float sdata[256];
  float local = 0.0f;
  for (int i = threadIdx.x; i < NROWS; i += 256) {
    const float neg1 = R[i]         - EXP5 + R[2 * NROWS + i];
    const float neg2 = R[NROWS + i] - EXP5 + R[3 * NROWS + i];
    local += -d12[i] * INV_TAU + 0.5f * (__logf(neg1) + __logf(neg2));
  }
  sdata[threadIdx.x] = local;
  __syncthreads();
  #pragma unroll
  for (int st = 128; st > 0; st >>= 1) {
    if (threadIdx.x < st) sdata[threadIdx.x] += sdata[threadIdx.x + st];
    __syncthreads();
  }
  if (threadIdx.x == 0) out[0] = sdata[0] * (1.0f / (float)NROWS);
}

// ---------------------------------------------------------------------------
extern "C" void kernel_launch(void* const* d_in, const int* in_sizes, int n_in,
                              void* d_out, int out_size, void* d_ws, size_t ws_size,
                              hipStream_t stream) {
  const float* h1 = (const float*)d_in[0];
  const float* h2 = (const float*)d_in[1];

  char* ws = (char*)d_ws;
  _Float16* h1n = (_Float16*)ws;                                   // 2 MB
  _Float16* h2n = (_Float16*)(ws + (size_t)NROWS * DIM * 2);       // 2 MB
  float*    R   = (float*)(ws + 2 * (size_t)NROWS * DIM * 2);      // 4*8192 f32
  float*    d12 = R + 4 * NROWS;                                   // 8192 f32

  zero_kernel<<<(4 * NROWS + 255) / 256, 256, 0, stream>>>(R, 4 * NROWS);
  norm_kernel<<<NROWS / 8, 256, 0, stream>>>(h1, h2, h1n, h2n, d12);

  dim3 grid(CSPLIT, NROWS / WGROWS, 3);    // 8 x 32 x 3 = 768 blocks
  gram_kernel<<<grid, 256, 0, stream>>>(h1n, h2n, R);

  loss_kernel<<<1, 256, 0, stream>>>(R, d12, (float*)d_out);
}
